// Wavelet1D_8280696947252
// MI455X (gfx1250) — matrix operand order repaired and checked by reading, not compiled, not run on hardware
//
#include <hip/hip_runtime.h>
#include <stdint.h>

typedef __attribute__((ext_vector_type(16))) _Float16 v16h;
typedef __attribute__((ext_vector_type(8)))  float    v8f;
typedef __attribute__((ext_vector_type(2)))  float    v2f;
typedef __attribute__((ext_vector_type(4)))  int      v4i;

#define TLEN  4096
#define NPAD  8192
#define BATCH 64
#define KF    4096                      // positive frequencies used: k = 1..4096
#define PHI   7.6699039394282058e-4f    // 2*pi/8192

// ---- CDNA5 async global->LDS copy (ASYNCcnt path) --------------------------
__device__ __forceinline__ void async_copy_b128(const void* g, void* l) {
#if __has_builtin(__builtin_amdgcn_global_load_async_to_lds_b128)
  __builtin_amdgcn_global_load_async_to_lds_b128((v4i*)g, (v4i*)l, 0, 0);
#else
  unsigned int loff = (unsigned int)(uintptr_t)l;   // LDS byte address (low 32b)
  asm volatile("global_load_async_to_lds_b128 %0, %1, off"
               :: "v"(loff), "v"(g) : "memory");
#endif
}

__device__ __forceinline__ void wait_async0() {
#if __has_builtin(__builtin_amdgcn_s_wait_asynccnt)
  __builtin_amdgcn_s_wait_asynccnt(0);
#else
  asm volatile("s_wait_asynccnt 0x0" ::: "memory");
#endif
}

// ---------------- kernel 0: reflect pad (64x4096 -> 64x8192) ----------------
__global__ void pad_kernel(const float* __restrict__ x, float* __restrict__ xp) {
  int idx = blockIdx.x * blockDim.x + threadIdx.x;
  if (idx >= BATCH * NPAD) return;
  int b = idx >> 13, j = idx & (NPAD - 1);
  int src;
  if (j < 2048)       src = 2047 - j;      // flip(x[:, :2048])
  else if (j < 6144)  src = j - 2048;      // x
  else                src = 10239 - j;     // flip(x[:, -2048:])
  xp[idx] = x[b * TLEN + src];
}

// ---------------- kernel 1: forward DFT via f32 WMMA ------------------------
// F[b,k] = sum_n xp[b,n] * e^{-2pi i k n / 8192}, k = kk+1, kk in [0,4096)
__global__ void __launch_bounds__(128) stage1_kernel(const float* __restrict__ xp,
                                                     float* __restrict__ Fre,
                                                     float* __restrict__ Fim) {
  int lane = threadIdx.x & 31;
  int wave = threadIdx.x >> 5;
  int wg   = blockIdx.x * 4 + wave;
  int b0   = (wg & 3) * 16;
  int kf0  = (wg >> 2) * 16;
  int lh   = lane >> 4;
  int ll   = lane & 15;
  int kfreq = kf0 + ll + 1;

  float bc[2], bs[2];
#pragma unroll
  for (int v = 0; v < 2; ++v) {
    int q = (kfreq * (lh * 2 + v)) & 8191;         // exact integer phase mod 8192
    __sincosf(q * PHI, &bs[v], &bc[v]);
  }
  int dq = (kfreq * 4) & 8191;                     // n advances by 4 per step
  float rs, rc;
  __sincosf(dq * PHI, &rs, &rc);
  { float inv = __frsqrt_rn(rc * rc + rs * rs); rc *= inv; rs *= inv; }

  v8f accC = {}; v8f accS = {};
  const float* arow = xp + (size_t)(b0 + ll) * NPAD + lh * 2;

  for (int n0 = 0; n0 < NPAD; n0 += 4) {
    v2f a = *(const v2f*)(arow + n0);              // A: K = lh*2 + {0,1}
    v2f bcv; bcv[0] = bc[0]; bcv[1] = bc[1];
    v2f bsv; bsv[0] = bs[0]; bsv[1] = bs[1];
    accC = __builtin_amdgcn_wmma_f32_16x16x4_f32(false, a, false, bcv, (short)0, accC, false, false);
    accS = __builtin_amdgcn_wmma_f32_16x16x4_f32(false, a, false, bsv, (short)0, accS, false, false);
#pragma unroll
    for (int v = 0; v < 2; ++v) {
      float c = bc[v] * rc - bs[v] * rs;
      float s = bc[v] * rs + bs[v] * rc;
      bc[v] = c; bs[v] = s;
    }
  }
#pragma unroll
  for (int v = 0; v < 8; ++v) {                    // C/D layout: M = v + lh*8
    int row = v + lh * 8;
    int o = (b0 + row) * KF + kf0 + ll;
    Fre[o] =  accC[v];
    Fim[o] = -accS[v];
  }
}

// ---------------- kernel 2: G[b,s,k] = F[b,k] * wft[s,k] / N  (f16) ---------
__global__ void gmul_kernel(const float* __restrict__ Fre, const float* __restrict__ Fim,
                            const float* __restrict__ wft,
                            _Float16* __restrict__ Gre, _Float16* __restrict__ Gim, int S) {
  int idx = blockIdx.x * blockDim.x + threadIdx.x;
  if (idx >= BATCH * S * KF) return;
  int kk = idx & (KF - 1);
  int r  = idx >> 12;
  int b  = r / S, s = r - b * S;
  float w = wft[s * NPAD + kk + 1] * (1.0f / 8192.0f);   // fold IFFT 1/N scale
  Gre[idx] = (_Float16)(Fre[b * KF + kk] * w);
  Gim[idx] = (_Float16)(Fim[b * KF + kk] * w);
}

// ---------------- kernel 3: inverse DFT + |.| + log via f16 WMMA ------------
// y[r,m] = sum_kk G[r,kk] * e^{+2pi i (kk+1)(m+2048)/8192};  out = 0.5*log(|y|^2)
// WG = 256 thr (8 waves): 32 rows x 128 cols; per wave 2 row-frags x 1 col-frag.
// Double-buffered LDS filled by GLOBAL_LOAD_ASYNC_TO_LDS_B128 (ASYNCcnt).
__global__ void __launch_bounds__(256) stage2_kernel(const _Float16* __restrict__ Gre,
                                                     const _Float16* __restrict__ Gim,
                                                     float* __restrict__ out) {
  __shared__ _Float16 ldsA[2][2][32][32];  // [buf][plane re/im][row][K] = 8 KB
  int tid  = threadIdx.x;
  int lane = tid & 31, wave = tid >> 5;
  int lh   = lane >> 4, ll = lane & 15;
  int r0 = blockIdx.x * 32;
  int m0 = blockIdx.y * 128 + wave * 16;
  int m  = m0 + ll;

  // B (twiddle) recurrence: per lane 16 slots, slotK = lh*16 + j (same K order as the A reads from LDS)
  float er[16], ei[16];
#pragma unroll
  for (int j = 0; j < 16; ++j) {
    int q = ((lh * 16 + j + 1) * (m + 2048)) & 8191;   // exact integer phase
    __sincosf(q * PHI, &ei[j], &er[j]);
  }
  int dq = (32 * (m + 2048)) & 8191;                   // kk advances by 32/step
  float rs, rc;
  __sincosf(dq * PHI, &rs, &rc);
  { float inv = __frsqrt_rn(rc * rc + rs * rs); rc *= inv; rs *= inv; }

  v8f aRR0 = {}, aII0 = {}, aRI0 = {}, aIR0 = {};
  v8f aRR1 = {}, aII1 = {}, aRI1 = {}, aIR1 = {};

  // async loader mapping: each thread moves one 16B chunk per K-step
  int lplane = tid >> 7;           // 0: re, 1: im
  int lrow   = (tid & 127) >> 2;   // 0..31
  int lseg   = tid & 3;            // 16B segment within 64B row
  const char* gbase = lplane ? (const char*)Gim : (const char*)Gre;
  const char* grow  = gbase + (size_t)(r0 + lrow) * (KF * 2) + lseg * 16;

#define ISSUE_TILE(kb, buf)                                                   \
  async_copy_b128(grow + (size_t)(kb) * 2,                                    \
                  (void*)&ldsA[(buf)][lplane][lrow][lseg * 8])

  ISSUE_TILE(0, 0);

  for (int step = 0; step < KF / 32; ++step) {
    int buf = step & 1;
    wait_async0();
    __syncthreads();
    if (step + 1 < KF / 32) ISSUE_TILE((step + 1) * 32, buf ^ 1);

    // pack B fragments (f16) from f32 recurrence state, then rotate
    v16h ere, eim;
#pragma unroll
    for (int j = 0; j < 16; ++j) { ere[j] = (_Float16)er[j]; eim[j] = (_Float16)ei[j]; }
#pragma unroll
    for (int j = 0; j < 16; ++j) {
      float c = er[j] * rc - ei[j] * rs;
      float s = er[j] * rs + ei[j] * rc;
      er[j] = c; ei[j] = s;
    }

    // A fragments from LDS: row M=ll (+16 for second tile), K chunks lh*16, lh*16+8
    union AF { v16h h; float4 q[2]; } gr0, gi0, gr1, gi1;
    gr0.q[0] = *(const float4*)&ldsA[buf][0][ll][lh * 16];
    gr0.q[1] = *(const float4*)&ldsA[buf][0][ll][lh * 16 + 8];
    gi0.q[0] = *(const float4*)&ldsA[buf][1][ll][lh * 16];
    gi0.q[1] = *(const float4*)&ldsA[buf][1][ll][lh * 16 + 8];
    gr1.q[0] = *(const float4*)&ldsA[buf][0][16 + ll][lh * 16];
    gr1.q[1] = *(const float4*)&ldsA[buf][0][16 + ll][lh * 16 + 8];
    gi1.q[0] = *(const float4*)&ldsA[buf][1][16 + ll][lh * 16];
    gi1.q[1] = *(const float4*)&ldsA[buf][1][16 + ll][lh * 16 + 8];

    aRR0 = __builtin_amdgcn_wmma_f32_16x16x32_f16(false, gr0.h, false, ere, (short)0, aRR0, false, false);
    aIR0 = __builtin_amdgcn_wmma_f32_16x16x32_f16(false, gi0.h, false, ere, (short)0, aIR0, false, false);
    aRI0 = __builtin_amdgcn_wmma_f32_16x16x32_f16(false, gr0.h, false, eim, (short)0, aRI0, false, false);
    aII0 = __builtin_amdgcn_wmma_f32_16x16x32_f16(false, gi0.h, false, eim, (short)0, aII0, false, false);
    aRR1 = __builtin_amdgcn_wmma_f32_16x16x32_f16(false, gr1.h, false, ere, (short)0, aRR1, false, false);
    aIR1 = __builtin_amdgcn_wmma_f32_16x16x32_f16(false, gi1.h, false, ere, (short)0, aIR1, false, false);
    aRI1 = __builtin_amdgcn_wmma_f32_16x16x32_f16(false, gr1.h, false, eim, (short)0, aRI1, false, false);
    aII1 = __builtin_amdgcn_wmma_f32_16x16x32_f16(false, gi1.h, false, eim, (short)0, aII1, false, false);
  }
#undef ISSUE_TILE

#pragma unroll
  for (int v = 0; v < 8; ++v) {
    int row = v + lh * 8;
    float yre = aRR0[v] - aII0[v];
    float yim = aRI0[v] + aIR0[v];
    out[(size_t)(r0 + row) * TLEN + m] = 0.5f * __logf(yre * yre + yim * yim);
    yre = aRR1[v] - aII1[v];
    yim = aRI1[v] + aIR1[v];
    out[(size_t)(r0 + 16 + row) * TLEN + m] = 0.5f * __logf(yre * yre + yim * yim);
  }
}

extern "C" void kernel_launch(void* const* d_in, const int* in_sizes, int n_in,
                              void* d_out, int out_size, void* d_ws, size_t ws_size,
                              hipStream_t stream) {
  const float* x   = (const float*)d_in[0];
  const float* wft = (const float*)d_in[1];
  int S = in_sizes[1] / NPAD;                 // 56 scales

  float* xp  = (float*)d_ws;                  // 64*8192 f32      (2 MB)
  float* Fre = xp  + BATCH * NPAD;            // 64*4096 f32      (1 MB)
  float* Fim = Fre + BATCH * KF;              // 64*4096 f32      (1 MB)
  _Float16* Gre = (_Float16*)(Fim + BATCH * KF);          // R*4096 f16 (29.4 MB)
  _Float16* Gim = Gre + (size_t)BATCH * S * KF;           // R*4096 f16 (29.4 MB)
  float* out = (float*)d_out;

  pad_kernel<<<(BATCH * NPAD + 255) / 256, 256, 0, stream>>>(x, xp);
  stage1_kernel<<<(4 * (KF / 16)) / 4, 128, 0, stream>>>(xp, Fre, Fim);
  int total = BATCH * S * KF;
  gmul_kernel<<<(total + 255) / 256, 256, 0, stream>>>(Fre, Fim, wft, Gre, Gim, S);
  dim3 g2((BATCH * S) / 32, TLEN / 128);
  stage2_kernel<<<g2, 256, 0, stream>>>(Gre, Gim, out);
}
